// SFIModel_pipeline2_8478265442835
// MI455X (gfx1250) — matrix operand order repaired and checked by reading, not compiled, not run on hardware
//
#include <hip/hip_runtime.h>
#include <hip/hip_bf16.h>
#include <math.h>

// ---------------------------------------------------------------------------
// SFI pipeline for MI455X (gfx1250, wave32, WMMA)
// All matmul-shaped work mapped to v_wmma_f32_16x16x32_bf16.
// ---------------------------------------------------------------------------

typedef __bf16 bf16_t;
typedef bf16_t v16bf __attribute__((ext_vector_type(16)));
typedef float  v8f   __attribute__((ext_vector_type(8)));

#define WMMA_BF16(a, b, acc) \
  __builtin_amdgcn_wmma_f32_16x16x32_bf16(false, (a), false, (b), (short)0, (acc), false, false)

// Problem constants (from reference)
constexpr int Bb   = 32;
constexpr int CDD  = 5;
constexpr int HIS  = 100;
constexpr int Ss   = 20;
constexpr int Ll   = 3;
constexpr int Ff   = 150;
constexpr int Kk   = 18;
constexpr int RDIM = 256;
constexpr int EMB  = Ss * Ll * Ff;       // 9000
constexpr int NBC  = Bb * CDD;           // 160

// Workspace layout (bytes, 256-aligned)
constexpr size_t WS_HID  = 0;                                  // 2880 ints
constexpr size_t WS_X    = 11520;                              // 160*3*18*400 bf16 = 6,912,000 B
constexpr size_t WS_P1   = WS_X + (size_t)NBC*Ll*18*400*2;     // 160*32*216 bf16 = 2,211,840 B
constexpr size_t WS_FEAT = WS_P1 + (size_t)NBC*32*216*2;       // 160*128 f32
// total ~9.2 MB

// ---- WMMA fragment loaders (from LDS, row-major [row][ldk] bf16) ----------
// A fragment (16x32, 16-bit): lanes 0-15 row M, halves K {kb..kb+7, kb+16..kb+23}
// with kb = kk + 8*(lane>=16).
__device__ __forceinline__ v16bf frag_a(const bf16_t* A, int ldk, int row, int kk, int lane) {
  const bf16_t* p = A + row * ldk + kk + ((lane >> 4) & 1) * 8;
  v16bf a;
#pragma unroll
  for (int i = 0; i < 8; ++i) { a[i] = p[i]; a[8 + i] = p[16 + i]; }
  return a;
}
// B fragment (32x16, 16-bit, stored as [N][K]): lanes col N, halves K {kb..kb+7, kb+16..kb+23}
// with kb = kk + 8*(lane>=16) (same k order as frag_a).
__device__ __forceinline__ v16bf frag_b(const bf16_t* Bm, int ldk, int col, int kk, int lane) {
  const bf16_t* p = Bm + col * ldk + kk + ((lane >> 4) & 1) * 8;
  v16bf b;
#pragma unroll
  for (int i = 0; i < 8; ++i) { b[i] = p[i]; b[8 + i] = p[16 + i]; }
  return b;
}

// ---------------------------------------------------------------------------
// K1: attention (bcr,bhr->bch) + top-18 (stable descending => strict '>')
// ---------------------------------------------------------------------------
__global__ void k1_attn_topk(const int* __restrict__ cdd_id, const int* __restrict__ his_id,
                             const float* __restrict__ repr, int* __restrict__ hid_sel) {
  int blk = blockIdx.x;
  int b = blk / CDD, c = blk % CDD;
  __shared__ float cddv[RDIM];
  __shared__ float attnv[HIS];
  int tid = threadIdx.x;
  const float* crow = repr + (size_t)cdd_id[b * CDD + c] * RDIM;
  for (int i = tid; i < RDIM; i += blockDim.x) cddv[i] = crow[i];
  __syncthreads();
  if (tid < HIS) {
    const float* hrow = repr + (size_t)his_id[b * HIS + tid] * RDIM;
    float s = 0.f;
    for (int f = 0; f < RDIM; ++f) s += cddv[f] * hrow[f];
    attnv[tid] = s;
  }
  __syncthreads();
  if (tid == 0) {
    for (int k = 0; k < Kk; ++k) {
      float best = -INFINITY; int bi = 0;
      for (int h = 0; h < HIS; ++h)
        if (attnv[h] > best) { best = attnv[h]; bi = h; }
      hid_sel[(b * CDD + c) * Kk + k] = his_id[b * HIS + bi];
      attnv[bi] = -INFINITY;
    }
  }
}

// ---------------------------------------------------------------------------
// K2: fusion GEMM per (bc,k,l): out[s,t] = sum_f A[s,f]*B[t,f] / sqrt(F)
// 20x20x150 padded to 32x32x160 -> 4 waves x (one 16x16 tile, 5 WMMA).
// Output written in conv1 NCDHW layout (bc, l, k, s, t) as bf16.
// ---------------------------------------------------------------------------
__global__ void k2_fusion(const int* __restrict__ cdd_id, const int* __restrict__ hid_sel,
                          const float* __restrict__ emb, bf16_t* __restrict__ xo) {
  int blk = blockIdx.x;
  int l  = blk % Ll;
  int k  = (blk / Ll) % Kk;
  int bc = blk / (Ll * Kk);
  const float* crow = emb + (size_t)cdd_id[bc] * EMB;
  const float* hrow = emb + (size_t)hid_sel[bc * Kk + k] * EMB;
  __builtin_prefetch(hrow, 0, 1);   // global_prefetch_b8

  __shared__ bf16_t sA[32 * 160];
  __shared__ bf16_t sB[32 * 160];
  int tid = threadIdx.x;
  for (int idx = tid; idx < 32 * 160; idx += 128) {
    int s = idx / 160, f = idx % 160;
    bool v = (s < Ss) && (f < Ff);
    sA[idx] = v ? (bf16_t)crow[(s * Ll + l) * Ff + f] : (bf16_t)0.f;
    sB[idx] = v ? (bf16_t)hrow[(s * Ll + l) * Ff + f] : (bf16_t)0.f;
  }
  __syncthreads();

  int wave = tid >> 5, lane = tid & 31, hi = (lane >> 4) & 1;
  int m0 = (wave & 1) * 16, n0 = (wave >> 1) * 16;
  v8f acc = {};
#pragma unroll
  for (int ks = 0; ks < 5; ++ks) {
    v16bf a  = frag_a(sA, 160, m0 + (lane & 15), ks * 32, lane);
    v16bf bm = frag_b(sB, 160, n0 + (lane & 15), ks * 32, lane);
    acc = WMMA_BF16(a, bm, acc);
  }
  const float sc = 0.081649658092772603f;   // 1/sqrt(150)
#pragma unroll
  for (int r = 0; r < 8; ++r) {
    int s = m0 + r + hi * 8;
    int t = n0 + (lane & 15);
    if (s < Ss && t < Ss)
      xo[((size_t)(bc * Ll + l) * 18 + k) * 400 + s * 20 + t] = (bf16_t)(acc[r] * sc);
  }
}

// ---------------------------------------------------------------------------
// K3: conv1 (3->32ch, 3x3x3, pad 1) + bias + relu + maxpool3 (fused via LDS
// uint atomic-max; valid since post-relu >= 0).  Implicit GEMM:
// M=32 oc (2 tiles), K=81->96 (3 WMMA steps), N=16 positions per im2col tile.
// One block per (bc, pooled-z).  972 positions -> 61 tiles, 2 tiles/iter.
// ---------------------------------------------------------------------------
__global__ void k3_conv1(const bf16_t* __restrict__ xin, const float* __restrict__ w,
                         const float* __restrict__ bias, bf16_t* __restrict__ p1) {
  int bc = blockIdx.x / 6, pz = blockIdx.x % 6;
  __shared__ bf16_t in_s[3 * 5 * 22 * 22];   // [ci][dzi 0..4][yy 0..21][xx 0..21], zero halo
  __shared__ bf16_t w1[32 * 96];
  __shared__ bf16_t im[2][16 * 96];
  __shared__ unsigned pool[32 * 36];         // float bits, init 0
  __shared__ float biasS[32];
  int tid = threadIdx.x;

  for (int idx = tid; idx < 32 * 96; idx += 128) {
    int kq = idx % 96;
    w1[idx] = (kq < 81) ? (bf16_t)w[(idx / 96) * 81 + kq] : (bf16_t)0.f;
  }
  for (int idx = tid; idx < 3 * 5 * 22 * 22; idx += 128) {
    int ci = idx / 2420, rem = idx % 2420;
    int dzi = rem / 484, yy = (rem % 484) / 22, xx = rem % 22;
    int d = 3 * pz - 1 + dzi, y = yy - 1, x = xx - 1;
    bf16_t v = (bf16_t)0.f;
    if ((unsigned)d < 18u && (unsigned)y < 20u && (unsigned)x < 20u)
      v = xin[((size_t)(bc * Ll + ci) * 18 + d) * 400 + y * 20 + x];
    in_s[idx] = v;
  }
  for (int idx = tid; idx < 32 * 36; idx += 128) pool[idx] = 0u;
  if (tid < 32) biasS[tid] = bias[tid];
  __syncthreads();

  int wave = tid >> 5, lane = tid & 31, hi = (lane >> 4) & 1;
  int tg = wave >> 1;               // which of the 2 im2col buffers this wave consumes
  int m0 = (wave & 1) * 16;         // oc tile
  for (int it = 0; it < 31; ++it) {
    // build im2col for tiles 2*it and 2*it+1 (16 positions x 96 K each)
    for (int idx = tid; idx < 2 * 16 * 96; idx += 128) {
      int buf = idx / 1536, p = (idx % 1536) / 96, kq = idx % 96;
      int pos = (2 * it + buf) * 16 + p;
      bf16_t v = (bf16_t)0.f;
      if (pos < 972 && kq < 81) {
        int dd = pos / 324, y = (pos % 324) / 18, x2 = pos % 18;
        int ci = kq / 27, rr = kq % 27, dz = rr / 9, dy = (rr % 9) / 3, dx = rr % 3;
        v = in_s[((ci * 5 + dd + dz) * 22 + y + dy) * 22 + (x2 + dx)];
      }
      im[buf][p * 96 + kq] = v;
    }
    __syncthreads();
    int ti = 2 * it + tg;           // wave-uniform
    if (ti < 61) {
      v8f acc = {};
#pragma unroll
      for (int ks = 0; ks < 3; ++ks) {
        v16bf a  = frag_a(w1, 96, m0 + (lane & 15), ks * 32, lane);
        v16bf bm = frag_b(im[tg], 96, lane & 15, ks * 32, lane);
        acc = WMMA_BF16(a, bm, acc);
      }
      int pos = ti * 16 + (lane & 15);
      if (pos < 972) {
        int y = (pos % 324) / 18, x2 = pos % 18;
        int cell = (y / 3) * 6 + (x2 / 3);
#pragma unroll
        for (int r = 0; r < 8; ++r) {
          int oc = m0 + r + hi * 8;
          float val = fmaxf(acc[r] + biasS[oc], 0.f);
          atomicMax(&pool[oc * 36 + cell], __float_as_uint(val));
        }
      }
    }
    __syncthreads();
  }
  for (int idx = tid; idx < 32 * 36; idx += 128) {
    int oc = idx / 36, cell = idx % 36;
    p1[((size_t)(bc * 32 + oc) * 6 + pz) * 36 + cell] = (bf16_t)__uint_as_float(pool[idx]);
  }
}

// ---------------------------------------------------------------------------
// K4: conv2 (32->16ch, 3x3x3, pad 1, 6^3 spatial) + bias + relu + maxpool3.
// M=16 oc (1 tile), K=864 = 27 x 32 WMMA steps, N=16 positions/tile.
// Per-k-step im2col staging keeps LDS ~30 KB.  One block (2 waves) per sample.
// ---------------------------------------------------------------------------
__global__ void k4_conv2(const bf16_t* __restrict__ p1, const float* __restrict__ w,
                         const float* __restrict__ bias, float* __restrict__ feat) {
  int bc = blockIdx.x;
  __shared__ bf16_t w2[16 * 864];
  __shared__ bf16_t im2[2][16 * 32];
  __shared__ unsigned pool2[16 * 8];
  __shared__ float biasS[16];
  int tid = threadIdx.x;
  for (int idx = tid; idx < 16 * 864; idx += 64) w2[idx] = (bf16_t)w[idx];
  for (int idx = tid; idx < 128; idx += 64) pool2[idx] = 0u;
  if (tid < 16) biasS[tid] = bias[tid];
  __syncthreads();

  int wave = tid >> 5, lane = tid & 31, hi = (lane >> 4) & 1;
  const bf16_t* p1s = p1 + (size_t)bc * 32 * 216;
  for (int it = 0; it < 7; ++it) {     // 14 tiles of 16 positions (216 total)
    int ti = 2 * it + wave;            // wave-uniform
    v8f acc = {};
    for (int ks = 0; ks < 27; ++ks) {
      for (int idx = tid; idx < 1024; idx += 64) {
        int buf = idx >> 9, p = (idx >> 5) & 15, kk = idx & 31;
        int pos = (2 * it + buf) * 16 + p;
        int kq = ks * 32 + kk;
        int ci = kq / 27, rr = kq % 27, dz = rr / 9, dy = (rr % 9) / 3, dx = rr % 3;
        bf16_t v = (bf16_t)0.f;
        if (pos < 216) {
          int od = pos / 36, oy = (pos % 36) / 6, ox = pos % 6;
          int id = od + dz - 1, iy = oy + dy - 1, ix = ox + dx - 1;
          if ((unsigned)id < 6u && (unsigned)iy < 6u && (unsigned)ix < 6u)
            v = p1s[(ci * 6 + id) * 36 + iy * 6 + ix];
        }
        im2[buf][p * 32 + kk] = v;
      }
      __syncthreads();
      if (ti < 14) {
        v16bf a  = frag_a(w2, 864, lane & 15, ks * 32, lane);
        v16bf bm = frag_b(im2[wave], 32, lane & 15, 0, lane);
        acc = WMMA_BF16(a, bm, acc);
      }
      __syncthreads();
    }
    if (ti < 14) {
      int pos = ti * 16 + (lane & 15);
      if (pos < 216) {
        int od = pos / 36, oy = (pos % 36) / 6, ox = pos % 6;
        int cell = (od / 3) * 4 + (oy / 3) * 2 + (ox / 3);
#pragma unroll
        for (int r = 0; r < 8; ++r) {
          int oc = r + hi * 8;
          float val = fmaxf(acc[r] + biasS[oc], 0.f);
          atomicMax(&pool2[oc * 8 + cell], __float_as_uint(val));
        }
      }
    }
  }
  __syncthreads();
  for (int idx = tid; idx < 128; idx += 64)
    feat[(size_t)bc * 128 + idx] = __uint_as_float(pool2[idx]);  // (oc, d, h, w) order
}

// ---------------------------------------------------------------------------
// K5: score = feat . ltr_w + b ; log_softmax over CDD
// ---------------------------------------------------------------------------
__global__ void k5_score(const float* __restrict__ feat, const float* __restrict__ lw,
                         const float* __restrict__ lb, float* __restrict__ out) {
  __shared__ float sc[NBC];
  int tid = threadIdx.x;
  if (tid < NBC) {
    const float* fr = feat + (size_t)tid * 128;
    float s = 0.f;
    for (int j = 0; j < 128; ++j) s += fr[j] * lw[j];
    sc[tid] = s + lb[0];
  }
  __syncthreads();
  if (tid < Bb) {
    float m = -INFINITY;
    for (int c = 0; c < CDD; ++c) m = fmaxf(m, sc[tid * CDD + c]);
    float se = 0.f;
    for (int c = 0; c < CDD; ++c) se += expf(sc[tid * CDD + c] - m);
    float lse = m + logf(se);
    for (int c = 0; c < CDD; ++c) out[tid * CDD + c] = sc[tid * CDD + c] - lse;
  }
}

// ---------------------------------------------------------------------------
extern "C" void kernel_launch(void* const* d_in, const int* in_sizes, int n_in,
                              void* d_out, int out_size, void* d_ws, size_t ws_size,
                              hipStream_t stream) {
  const int*   cdd_id = (const int*)d_in[0];
  const int*   his_id = (const int*)d_in[1];
  const float* repr   = (const float*)d_in[2];
  const float* emb    = (const float*)d_in[3];
  const float* c1w    = (const float*)d_in[4];
  const float* c1b    = (const float*)d_in[5];
  const float* c2w    = (const float*)d_in[6];
  const float* c2b    = (const float*)d_in[7];
  const float* lw     = (const float*)d_in[8];
  const float* lb     = (const float*)d_in[9];
  float*       outp   = (float*)d_out;

  char* ws = (char*)d_ws;
  int*    hid_sel = (int*)   (ws + WS_HID);
  bf16_t* xbuf    = (bf16_t*)(ws + WS_X);
  bf16_t* p1buf   = (bf16_t*)(ws + WS_P1);
  float*  featbuf = (float*) (ws + WS_FEAT);

  k1_attn_topk<<<NBC, 128, 0, stream>>>(cdd_id, his_id, repr, hid_sel);
  k2_fusion  <<<NBC * Kk * Ll, 128, 0, stream>>>(cdd_id, hid_sel, emb, xbuf);
  k3_conv1   <<<NBC * 6, 128, 0, stream>>>(xbuf, c1w, c1b, p1buf);
  k4_conv2   <<<NBC, 64, 0, stream>>>(p1buf, c2w, c2b, featbuf);
  k5_score   <<<1, 192, 0, stream>>>(featbuf, lw, lb, outp);
}
